// newRSNN_36223754174481
// MI455X (gfx1250) — matrix operand order repaired and checked by reading, not compiled, not run on hardware
//
#include <hip/hip_runtime.h>
#include <hip/hip_bf16.h>

typedef __attribute__((ext_vector_type(16))) __bf16 v16bf;
typedef __attribute__((ext_vector_type(8)))  __bf16 v8bf;
typedef __attribute__((ext_vector_type(8)))  float  v8f;
typedef __attribute__((ext_vector_type(2)))  int    v2i_t;

#define AS1 __attribute__((address_space(1)))
#define AS3 __attribute__((address_space(3)))

#define ALPHA   0.9f
#define RHO     0.985f
#define BETA_A  1.8f
#define TH      1.0f

#define KC 32            // K per WMMA chunk
#define NBUF 4           // LDS chunk ring depth
#define NBLK 32          // persistent grid size (8 N-blocks x 4 M-blocks)

#if defined(__has_builtin)
#if __has_builtin(__builtin_amdgcn_global_load_async_to_lds_b64) && \
    __has_builtin(__builtin_amdgcn_s_wait_asynccnt)
#define ASYNC_LDS 1
#endif
#endif
#ifndef ASYNC_LDS
#define ASYNC_LDS 0
#endif

static __device__ __forceinline__ v16bf pack16(v8bf lo, v8bf hi) {
    v16bf r;
#pragma unroll
    for (int i = 0; i < 8; ++i) { r[i] = lo[i]; r[i + 8] = hi[i]; }
    return r;
}

static __device__ __forceinline__ v8f wmma_bf16(v16bf a, v16bf b, v8f c) {
    return __builtin_amdgcn_wmma_f32_16x16x32_bf16(
        false, a, false, b, (short)0, c, false, false);
}

// One GEMM stage with fused epilogue. All geometry is compile-time.
// EPI: 0 = LIF soft reset, 1 = adaptive threshold + b_a decay, 2 = linear out
// C[m,n] = sum_k A[m,k] * W[n,k], A = concat(A0 | A1) split at KSPLIT.
// Block: 8 waves; wave w owns N strip n0 = bx*128 + w*16; 2 M-tiles (m0 = by*32).
// A chunks stream through a 4-deep LDS ring via async global->LDS copies with
// two fills in flight (wait_asynccnt(1) per iteration).
template <int EPI, int LDA0, int LDA1, int KSPLIT, int K, int N>
static __device__ __forceinline__ void gemm_stage(
    __bf16 (&lds)[NBUF][32][KC],
    const __bf16* __restrict__ A0,
    const __bf16* __restrict__ A1,
    const __bf16* __restrict__ W,
    const float* __restrict__ bias,
    float* __restrict__ vstate,
    float* __restrict__ ba,
    __bf16* __restrict__ sout,
    float* __restrict__ oout,
    int bx, int by)
{
    constexpr int NK = K / KC;

    const int tid  = threadIdx.x;
    const int lane = tid & 31;
    const int wave = tid >> 5;
    const int nsel = lane & 15;
    const int ksel = lane >> 4;
    const int m0   = by * 32;
    const int n0   = bx * 128 + wave * 16;
    const int n    = n0 + nsel;

    const __bf16* wrow = W + (size_t)n * K + ksel * 8;
    const float bn = bias[n];          // hoisted: latency hidden under K loop

    v8f acc[2] = {v8f{}, v8f{}};

    // cooperative A-chunk fill: 256 threads x 8B = 32 rows x 32 cols bf16
    auto fill = [&](int buf, int kblk) {
        const int row = tid >> 3;
        const int col = (tid & 7) << 2;
        const __bf16* src;
        if (kblk < KSPLIT)   // scalar-uniform: KC-chunks never straddle KSPLIT
            src = A0 + (size_t)(m0 + row) * LDA0 + (kblk + col);
        else
            src = A1 + (size_t)(m0 + row) * LDA1 + (kblk - KSPLIT + col);
        __bf16* dst = &lds[buf][row][col];
#if ASYNC_LDS
        __builtin_amdgcn_global_load_async_to_lds_b64(
            (AS1 v2i_t*)(AS1 const void*)src,
            (AS3 v2i_t*)(AS3 void*)dst, 0, 0);
#else
        *(unsigned long long*)dst = *(const unsigned long long*)src;
#endif
    };

    fill(0, 0);
    if (NK > 1) fill(1, KC);

#pragma clang loop unroll_count(2)
    for (int kc = 0; kc < NK; ++kc) {
#if ASYNC_LDS
        // async loads complete in order: <=1 outstanding => fill(kc) is done
        if (kc + 1 < NK) __builtin_amdgcn_s_wait_asynccnt(1);
        else             __builtin_amdgcn_s_wait_asynccnt(0);
#endif
        __syncthreads();

        const int buf = kc & (NBUF - 1);

        const __bf16* wp = wrow + kc * KC;
        __builtin_prefetch((const void*)(wp + 4 * KC), 0, 1);  // global_prefetch_b8
        v8bf blo = *(const v8bf*)(wp);
        v8bf bhi = *(const v8bf*)(wp + 16);
        v16bf bfrag = pack16(blo, bhi);

        v8bf a0lo = *(const v8bf*)(&lds[buf][nsel][ksel * 8]);
        v8bf a0hi = *(const v8bf*)(&lds[buf][nsel][16 + ksel * 8]);
        v8bf a1lo = *(const v8bf*)(&lds[buf][16 + nsel][ksel * 8]);
        v8bf a1hi = *(const v8bf*)(&lds[buf][16 + nsel][16 + ksel * 8]);

        acc[0] = wmma_bf16(pack16(a0lo, a0hi), bfrag, acc[0]);
        acc[1] = wmma_bf16(pack16(a1lo, a1hi), bfrag, acc[1]);

        // keep two fills in flight; buffer (kc+2)&3 was last read at kc-2
        if (kc + 2 < NK) fill((kc + 2) & (NBUF - 1), (kc + 2) * KC);
    }

#pragma unroll
    for (int mt = 0; mt < 2; ++mt) {
#pragma unroll
        for (int r = 0; r < 8; ++r) {
            const int m = m0 + mt * 16 + r + ksel * 8;
            const size_t idx = (size_t)m * N + n;
            const float a = acc[mt][r] + bn;
            if constexpr (EPI == 0) {
                float v = ALPHA * vstate[idx] + a;
                float s = (v - TH) > 0.0f ? 1.0f : 0.0f;
                v -= s * TH;
                vstate[idx] = v;
                sout[idx] = (__bf16)s;
            } else if constexpr (EPI == 1) {
                float v   = ALPHA * vstate[idx] + a;
                float bav = ba[idx];
                float th  = TH + BETA_A * bav;
                float s   = (v - th) > 0.0f ? 1.0f : 0.0f;
                v -= s * th;
                vstate[idx] = v;
                ba[idx] = RHO * bav + s;
                sout[idx] = (__bf16)s;
            } else {
                oout[idx] = a;
            }
        }
    }
}

// Persistent fused RSNN: one launch runs the whole T x SIM chain with a
// device-wide phase barrier between dependent GEMM stages.
__global__ __launch_bounds__(256) void rsnn_persist(
    const __bf16* __restrict__ inp_bf,
    const __bf16* __restrict__ Wpre, const __bf16* __restrict__ Wad,
    const __bf16* __restrict__ Wpost, const __bf16* __restrict__ Wout,
    const float* __restrict__ b_pre, const float* __restrict__ b_ad,
    const float* __restrict__ b_post, const float* __restrict__ b_out,
    float* __restrict__ v_pre, float* __restrict__ v_a,
    float* __restrict__ b_a, float* __restrict__ v_post,
    __bf16* __restrict__ spk, __bf16* __restrict__ s_pre,
    __bf16* __restrict__ s_post,
    float* __restrict__ out, unsigned* bar)
{
    constexpr int T = 32, B = 128, NIN = 512, NPRE = 1024, NAD = 1024,
                  NPOST = 1024, NOUT = 256, KX = NIN + NAD;

    __shared__ __align__(16) __bf16 lds[NBUF][32][KC];

    const int bid = blockIdx.x;      // 0..31
    const int bx  = bid & 7;
    const int by  = bid >> 3;

    unsigned ep = 0;
    auto gsync = [&]() {
        __syncthreads();
        __threadfence();             // release: state/spike stores visible at device scope
        ++ep;
        if (threadIdx.x == 0) {
            __hip_atomic_fetch_add(bar, 1u, __ATOMIC_RELEASE, __HIP_MEMORY_SCOPE_AGENT);
            while (__hip_atomic_load(bar, __ATOMIC_ACQUIRE, __HIP_MEMORY_SCOPE_AGENT)
                   < ep * (unsigned)NBLK)
                __builtin_amdgcn_s_sleep(2);
        }
        __syncthreads();
        __threadfence();             // acquire: drop stale near-cache lines
    };

    for (int t = 0; t < T; ++t) {
        const __bf16* xin = inp_bf + (size_t)t * B * NIN;
        for (int sim = 0; sim < 3; ++sim) {
            gemm_stage<0, NIN, NAD, NIN, KX, NPRE>(
                lds, xin, spk, Wpre, b_pre, v_pre, nullptr, s_pre, nullptr, bx, by);
            gsync();
            gemm_stage<1, NPRE, NPRE, NPRE, NPRE, NAD>(
                lds, s_pre, s_pre, Wad, b_ad, v_a, b_a, spk, nullptr, bx, by);
            gsync();
            gemm_stage<0, NIN, NAD, NIN, KX, NPOST>(
                lds, xin, spk, Wpost, b_post, v_post, nullptr, s_post, nullptr, bx, by);
            gsync();
            if (sim == 2) {
                if (bx < NOUT / 128)
                    gemm_stage<2, NPOST, NPOST, NPOST, NPOST, NOUT>(
                        lds, s_post, s_post, Wout, b_out, nullptr, nullptr, nullptr,
                        out + (size_t)t * B * NOUT, bx, by);
                gsync();
            }
        }
    }

    // tail: final spikes -> fp32 region of d_out (already globally synced)
    float* ospk = out + (size_t)T * B * NOUT;
    for (int i = bid * 256 + threadIdx.x; i < B * NAD; i += NBLK * 256)
        ospk[i] = (float)spk[i];
}

__global__ void cvt_f32_bf16(const float* __restrict__ s, __bf16* __restrict__ d, int n) {
    int i = blockIdx.x * blockDim.x + threadIdx.x;
    if (i < n) d[i] = (__bf16)s[i];
}

extern "C" void kernel_launch(void* const* d_in, const int* in_sizes, int n_in,
                              void* d_out, int out_size, void* d_ws, size_t ws_size,
                              hipStream_t stream) {
    (void)in_sizes; (void)n_in; (void)out_size; (void)ws_size;

    constexpr int T = 32, B = 128, NIN = 512, NPRE = 1024, NAD = 1024,
                  NPOST = 1024, NOUT = 256;
    constexpr int KX = NIN + NAD;

    const float* inp    = (const float*)d_in[0];
    const float* W_pre  = (const float*)d_in[1];
    const float* b_pre  = (const float*)d_in[2];
    const float* W_ad   = (const float*)d_in[3];
    const float* b_ad   = (const float*)d_in[4];
    const float* W_post = (const float*)d_in[5];
    const float* b_post = (const float*)d_in[6];
    const float* W_out  = (const float*)d_in[7];
    const float* b_out  = (const float*)d_in[8];
    float* out = (float*)d_out;

    char* p = (char*)d_ws;
    auto carve = [&](size_t bytes) -> void* {
        void* r = (void*)p;
        p += (bytes + 255) & ~(size_t)255;
        return r;
    };
    __bf16* inp_bf   = (__bf16*)carve((size_t)T * B * NIN * 2);
    __bf16* Wpre_bf  = (__bf16*)carve((size_t)NPRE * KX * 2);
    __bf16* Wad_bf   = (__bf16*)carve((size_t)NAD * NPRE * 2);
    __bf16* Wpost_bf = (__bf16*)carve((size_t)NPOST * KX * 2);
    __bf16* Wout_bf  = (__bf16*)carve((size_t)NOUT * NPOST * 2);
    __bf16* spk      = (__bf16*)carve((size_t)B * NAD * 2);
    __bf16* s_pre    = (__bf16*)carve((size_t)B * NPRE * 2);
    __bf16* s_post   = (__bf16*)carve((size_t)B * NPOST * 2);
    float*  v_pre    = (float*)carve((size_t)B * NPRE * 4);
    float*  v_a      = (float*)carve((size_t)B * NAD * 4);
    float*  b_a      = (float*)carve((size_t)B * NAD * 4);
    float*  v_post   = (float*)carve((size_t)B * NPOST * 4);
    unsigned* bar    = (unsigned*)carve(256);

    auto cvt = [&](const float* s, __bf16* d, int n) {
        cvt_f32_bf16<<<(n + 255) / 256, 256, 0, stream>>>(s, d, n);
    };
    cvt(inp,    inp_bf,   T * B * NIN);
    cvt(W_pre,  Wpre_bf,  NPRE * KX);
    cvt(W_ad,   Wad_bf,   NAD * NPRE);
    cvt(W_post, Wpost_bf, NPOST * KX);
    cvt(W_out,  Wout_bf,  NOUT * NPOST);

    (void)hipMemsetAsync(v_pre,  0, (size_t)B * NPRE * 4, stream);
    (void)hipMemsetAsync(v_a,    0, (size_t)B * NAD * 4, stream);
    (void)hipMemsetAsync(b_a,    0, (size_t)B * NAD * 4, stream);
    (void)hipMemsetAsync(v_post, 0, (size_t)B * NPOST * 4, stream);
    (void)hipMemsetAsync(spk,    0, (size_t)B * NAD * 2, stream);
    (void)hipMemsetAsync(bar,    0, 256, stream);

    rsnn_persist<<<dim3(NBLK), dim3(256), 0, stream>>>(
        inp_bf, Wpre_bf, Wad_bf, Wpost_bf, Wout_bf,
        b_pre, b_ad, b_post, b_out,
        v_pre, v_a, b_a, v_post,
        spk, s_pre, s_post,
        out, bar);
}
